// Net_53644141527184
// MI455X (gfx1250) — matrix operand order repaired and checked by reading, not compiled, not run on hardware
//
#include <hip/hip_runtime.h>
#include <hip/hip_bf16.h>
#include <stdint.h>

// ---------------------------------------------------------------------------
// feats gather + (768x384) x (384x25088) GEMM for MI455X (gfx1250, wave32).
// GEMM runs on v_wmma_f32_16x16x32_bf16 with a hi/lo bf16 split of fp32
// operands (Ahi*Bhi + Ahi*Blo + Alo*Bhi, fp32 accum) for ~fp32 accuracy at
// bf16-WMMA throughput. Gather is fused into the GEMM's B-tile LDS staging
// (double-buffered, stored in WMMA fragment order so fragment loads are two
// ds_load_b128 per plane), so every gathered element is read from HBM once.
// ---------------------------------------------------------------------------

typedef __attribute__((ext_vector_type(16))) __bf16          v16bf;
typedef __attribute__((ext_vector_type(16))) unsigned short  v16u;
typedef __attribute__((ext_vector_type(8)))  unsigned short  v8u;
typedef __attribute__((ext_vector_type(8)))  float           v8f;

#define C_IN 512
#define PP   256      // 16*16 positions
#define KDIM 384      // 3*128 gathered channels
#define MDIM 768      // output channels
#define NPB  196      // 14*14
#define NTOT 25088    // 128*196
#define NTILES (NTOT / 16)   // 1568
#define KSTEPS (KDIM / 32)   // 12

__device__ __forceinline__ unsigned short bf_hi(float f) {
  unsigned int u = __builtin_bit_cast(unsigned int, f);
  unsigned int r = u + 0x7FFFu + ((u >> 16) & 1u);   // round-to-nearest-even
  return (unsigned short)(r >> 16);
}
__device__ __forceinline__ float bf_f(unsigned short h) {
  unsigned int u = ((unsigned int)h) << 16;
  return __builtin_bit_cast(float, u);
}

// Kernel 0: split W (768x384 fp32, row-major) into bf16 hi/lo planes.
__global__ void wsplit_kernel(const float* __restrict__ W,
                              unsigned short* __restrict__ Whi,
                              unsigned short* __restrict__ Wlo, int n) {
  int i = blockIdx.x * blockDim.x + threadIdx.x;
  if (i >= n) return;
  float f = W[i];
  unsigned short h = bf_hi(f);
  unsigned short l = bf_hi(f - bf_f(h));
  Whi[i] = h;
  Wlo[i] = l;
}

// Kernel 1: one workgroup per 16-column N tile; 8 waves x 6 M-tiles = all M.
__global__ __launch_bounds__(256) void fused_gather_gemm(
    const float* __restrict__ x1, const float* __restrict__ x3,
    const float* __restrict__ x5,
    const unsigned short* __restrict__ Whi,
    const unsigned short* __restrict__ Wlo,
    const int* __restrict__ idx,          // (3,128,256) flat
    const int* __restrict__ offh,         // (3,128) flat
    const int* __restrict__ offw,         // (3,128) flat
    float* __restrict__ out)              // (128,768,196) flat
{
  // B tile staged in WMMA fragment order: lane L owns slots [L*16 .. L*16+15],
  // slot s of lane L = element (kk = (L>=16?16:0)+s, col = L&15). Double buffer.
  __shared__ unsigned short ldsBhi[2][32 * 16];
  __shared__ unsigned short ldsBlo[2][32 * 16];
  __shared__ int poff[KDIM];     // offh*16 + offw per gathered channel
  __shared__ int colp[16];       // h*16 + w per column
  __shared__ int colXB[16];      // b * C_IN * PP per column
  __shared__ int colOut[16];     // b*MDIM*NPB + hw per column

  const int tid = threadIdx.x;
  const int n0  = blockIdx.x * 16;

  // --- preamble: per-channel crop offsets, per-column bases ---
  for (int c = tid; c < KDIM; c += 256)
    poff[c] = offh[c] * 16 + offw[c];
  if (tid < 16) {
    int n  = n0 + tid;
    int b  = n / NPB;
    int hw = n - b * NPB;
    int h  = hw / 14;
    int w  = hw - h * 14;
    colp[tid]   = h * 16 + w;
    colXB[tid]  = b * (C_IN * PP);
    colOut[tid] = b * (MDIM * NPB) + hw;
  }

  const int lane  = tid & 31;
  const int wv    = tid >> 5;
  const int col   = lane & 15;
  const int khalf = (lane < 16) ? 0 : 16;   // A-matrix K base per lane half

  // Cooperative gather of one 32x16 B tile into fragment-order LDS.
  auto stage = [&](int k0, int buf) {
    #pragma unroll
    for (int e = tid; e < 512; e += 256) {
      int kk = e >> 4;
      int j  = e & 15;
      int c  = k0 + kk;
      int p  = poff[c] + colp[j];
      int ch = idx[c * PP + p];
      const float* xs = (c < 128) ? x1 : ((c < 256) ? x3 : x5);
      float v = xs[colXB[j] + ch * PP + p];
      unsigned short h = bf_hi(v);
      unsigned short l = bf_hi(v - bf_f(h));
      int L = j + ((kk >= 16) ? 16 : 0);
      int s = kk & 15;
      ldsBhi[buf][L * 16 + s] = h;
      ldsBlo[buf][L * 16 + s] = l;
    }
  };

  v8f acc[6] = {};

  __syncthreads();       // preamble tables visible
  stage(0, 0);           // prime buffer 0

  for (int ks = 0; ks < KSTEPS; ++ks) {
    const int cur = ks & 1;
    const int k0  = ks * 32;
    __syncthreads();     // buf[cur] stores done; buf[cur^1] readers done

    if (ks + 1 < KSTEPS) stage((ks + 1) * 32, cur ^ 1);  // overlap next gather

    // --- B fragments: two ds_load_b128 per plane, no repacking ---
    const v8u* ph = reinterpret_cast<const v8u*>(&ldsBhi[cur][lane * 16]);
    const v8u* pl = reinterpret_cast<const v8u*>(&ldsBlo[cur][lane * 16]);
    v8u bh0 = ph[0], bh1 = ph[1];
    v8u bl0 = pl[0], bl1 = pl[1];
    v16bf Bhi = __builtin_bit_cast(v16bf,
        __builtin_shufflevector(bh0, bh1, 0,1,2,3,4,5,6,7,8,9,10,11,12,13,14,15));
    v16bf Blo = __builtin_bit_cast(v16bf,
        __builtin_shufflevector(bl0, bl1, 0,1,2,3,4,5,6,7,8,9,10,11,12,13,14,15));

    // --- 6 M-tiles per wave: A slots = K khalf+{0..15} ---
    #pragma unroll
    for (int t = 0; t < 6; ++t) {
      int mrow = ((wv * 6 + t) * 16 + col) * KDIM + k0 + khalf;
      v8u h0 = *reinterpret_cast<const v8u*>(Whi + mrow);
      v8u h1 = *reinterpret_cast<const v8u*>(Whi + mrow + 8);
      v8u l0 = *reinterpret_cast<const v8u*>(Wlo + mrow);
      v8u l1 = *reinterpret_cast<const v8u*>(Wlo + mrow + 8);
      v16bf Ahi = __builtin_bit_cast(v16bf,
          __builtin_shufflevector(h0, h1, 0,1,2,3,4,5,6,7,8,9,10,11,12,13,14,15));
      v16bf Alo = __builtin_bit_cast(v16bf,
          __builtin_shufflevector(l0, l1, 0,1,2,3,4,5,6,7,8,9,10,11,12,13,14,15));

      acc[t] = __builtin_amdgcn_wmma_f32_16x16x32_bf16(
          false, Ahi, false, Bhi, (short)0, acc[t], false, false);
      acc[t] = __builtin_amdgcn_wmma_f32_16x16x32_bf16(
          false, Ahi, false, Blo, (short)0, acc[t], false, false);
      acc[t] = __builtin_amdgcn_wmma_f32_16x16x32_bf16(
          false, Alo, false, Bhi, (short)0, acc[t], false, false);
    }
  }

  // --- store: VGPR r -> M = r (lanes 0-15) or r+8 (lanes 16-31), N = col ---
  const int cOut = colOut[col];
  #pragma unroll
  for (int t = 0; t < 6; ++t) {
    int mbase = (wv * 6 + t) * 16 + ((lane < 16) ? 0 : 8);
    #pragma unroll
    for (int r = 0; r < 8; ++r)
      out[cOut + (mbase + r) * NPB] = acc[t][r];
  }
}

extern "C" void kernel_launch(void* const* d_in, const int* in_sizes, int n_in,
                              void* d_out, int out_size, void* d_ws, size_t ws_size,
                              hipStream_t stream) {
  const float* x1 = (const float*)d_in[0];
  const float* x3 = (const float*)d_in[1];
  const float* x5 = (const float*)d_in[2];
  const float* W  = (const float*)d_in[3];
  const int* idx  = (const int*)d_in[4];
  const int* offh = (const int*)d_in[5];
  const int* offw = (const int*)d_in[6];
  float* out = (float*)d_out;

  const int wn = MDIM * KDIM;                       // 294912
  unsigned short* Whi = (unsigned short*)d_ws;      // wn ushorts
  unsigned short* Wlo = Whi + wn;                   // wn ushorts (1.2 MB total)

  wsplit_kernel<<<(wn + 255) / 256, 256, 0, stream>>>(W, Whi, Wlo, wn);
  fused_gather_gemm<<<NTILES, 256, 0, stream>>>(x1, x3, x5, Whi, Wlo,
                                                idx, offh, offw, out);
}
